// _Experts_63007170232360
// MI455X (gfx1250) — matrix operand order repaired and checked by reading, not compiled, not run on hardware
//
#include <hip/hip_runtime.h>
#include <hip/hip_bf16.h>

// ---------------- problem constants ----------------
#define N_TOK   2048
#define HID     2048
#define INTER   768
#define NE      8
#define TOPK    2
#define ROWCAP  4352           // 136 tiles of 32 rows (>= 4096 + 8*31 padding)
#define MTILES  (ROWCAP / 32)  // 136

typedef _Float16 v4h  __attribute__((ext_vector_type(4)));
typedef _Float16 v8h  __attribute__((ext_vector_type(8)));
typedef _Float16 v16h __attribute__((ext_vector_type(16)));
typedef float    v8f  __attribute__((ext_vector_type(8)));

// Build a 16-half WMMA operand fragment from two 16-byte global loads.
__device__ __forceinline__ v16h ld_frag(const _Float16* p0, const _Float16* p1) {
    v8h a = *(const v8h*)p0;
    v8h b = *(const v8h*)p1;
    return __builtin_shufflevector(a, b, 0,1,2,3,4,5,6,7,8,9,10,11,12,13,14,15);
}

// ---------------- conversion fp32 -> fp16 ----------------
__global__ void k_f32_to_f16(const float* __restrict__ src, _Float16* __restrict__ dst, int n4) {
    int i = blockIdx.x * blockDim.x + threadIdx.x;
    if (i >= n4) return;
    float4 v = ((const float4*)src)[i];
    v4h h; h[0] = (_Float16)v.x; h[1] = (_Float16)v.y; h[2] = (_Float16)v.z; h[3] = (_Float16)v.w;
    ((v4h*)dst)[i] = h;
}

// ---------------- routing ----------------
__global__ void k_zero_small(int* counts, int* cursors, int* offs) {
    int t = threadIdx.x;
    if (t < NE)  { counts[t] = 0; cursors[t] = 0; }
    if (t <= NE) offs[t] = 0;
}

__global__ void k_fill_seg(int* seg) {
    int i = blockIdx.x * blockDim.x + threadIdx.x;
    if (i < ROWCAP) seg[i] = 0;   // sentinel token 0 for padded rows
}

__global__ void k_count(const int* __restrict__ topk, int* counts) {
    int i = blockIdx.x * blockDim.x + threadIdx.x;
    if (i >= N_TOK * TOPK) return;
    int e = topk[i];
    e = (e < 0) ? 0 : ((e >= NE) ? NE - 1 : e);
    atomicAdd(&counts[e], 1);
}

__global__ void k_scan(const int* __restrict__ counts, int* offs) {
    if (threadIdx.x == 0 && blockIdx.x == 0) {
        int acc = 0;
        for (int e = 0; e < NE; ++e) {
            offs[e] = acc;
            acc += (counts[e] + 31) & ~31;   // 32-align each expert segment
        }
        offs[NE] = acc;
    }
}

__global__ void k_scatter(const int* __restrict__ topk, const int* __restrict__ offs,
                          int* cursors, int* seg, int* pair_row) {
    int i = blockIdx.x * blockDim.x + threadIdx.x;
    if (i >= N_TOK * TOPK) return;
    int e = topk[i];
    e = (e < 0) ? 0 : ((e >= NE) ? NE - 1 : e);
    int slot = atomicAdd(&cursors[e], 1);
    int row  = offs[e] + slot;
    seg[row]    = i / TOPK;   // token id
    pair_row[i] = row;
}

// ---------------- GEMM1: gate_up projection + SiLU*up, f16 WMMA ----------------
// grid: (3, MTILES), block: 128 (4 waves).
// Wave computes a 32x64 gate tile + 32x64 up tile (2 M-frags x 4 N-frags each);
// both M-frags share every B fragment -> 2x B reuse vs M=16.
__global__ __launch_bounds__(128)
void moe_gemm1(const _Float16* __restrict__ Xh, const _Float16* __restrict__ Wgu,
               const int* __restrict__ seg, const int* __restrict__ offs,
               const int* __restrict__ counts, _Float16* __restrict__ act) {
    const int rowBase = blockIdx.y * 32;
    if (rowBase >= offs[NE]) return;
    int e = 0;
    #pragma unroll
    for (int i = 1; i < NE; ++i) if (rowBase >= offs[i]) e = i;
    if (rowBase >= offs[e] + counts[e]) return;   // pure-padding tile

    const int wave = threadIdx.x >> 5;
    const int lane = threadIdx.x & 31;
    const int half = lane >> 4;       // K sub-range select per ISA A/B layout
    const int r    = lane & 15;       // row (A) / column (B,D) within 16-tile
    const int colBase = blockIdx.x * 256 + wave * 64;

    const int tok0 = seg[rowBase + r];
    const int tok1 = seg[rowBase + 16 + r];
    const _Float16* aRow0 = Xh + (size_t)tok0 * HID;
    const _Float16* aRow1 = Xh + (size_t)tok1 * HID;

    const _Float16* bg[4];
    const _Float16* bu[4];
    #pragma unroll
    for (int t = 0; t < 4; ++t) {
        bg[t] = Wgu + ((size_t)e * (2 * INTER) + (colBase + t * 16 + r)) * HID + half * 8;
        bu[t] = bg[t] + (size_t)INTER * HID;
    }

    v8f accG[2][4] = {}; v8f accU[2][4] = {};
    for (int kk = 0; kk < HID; kk += 32) {
        v16h A0 = ld_frag(aRow0 + kk + half * 8, aRow0 + kk + 16 + half * 8);
        v16h A1 = ld_frag(aRow1 + kk + half * 8, aRow1 + kk + 16 + half * 8);
        #pragma unroll
        for (int t = 0; t < 4; ++t) {
            v16h B = ld_frag(bg[t] + kk, bg[t] + kk + 16);
            accG[0][t] = __builtin_amdgcn_wmma_f32_16x16x32_f16(false, A0, false, B,
                                                                (short)0, accG[0][t], false, false);
            accG[1][t] = __builtin_amdgcn_wmma_f32_16x16x32_f16(false, A1, false, B,
                                                                (short)0, accG[1][t], false, false);
        }
        #pragma unroll
        for (int t = 0; t < 4; ++t) {
            v16h B = ld_frag(bu[t] + kk, bu[t] + kk + 16);
            accU[0][t] = __builtin_amdgcn_wmma_f32_16x16x32_f16(false, A0, false, B,
                                                                (short)0, accU[0][t], false, false);
            accU[1][t] = __builtin_amdgcn_wmma_f32_16x16x32_f16(false, A1, false, B,
                                                                (short)0, accU[1][t], false, false);
        }
    }
    // D layout: acc[j] -> row (m*16 + j + half*8), col r (within 16-tile)
    #pragma unroll
    for (int m = 0; m < 2; ++m) {
        #pragma unroll
        for (int t = 0; t < 4; ++t) {
            #pragma unroll
            for (int j = 0; j < 8; ++j) {
                float g = accG[m][t][j];
                float u = accU[m][t][j];
                float s = g / (1.0f + __expf(-g));      // SiLU
                act[(size_t)(rowBase + m * 16 + j + half * 8) * INTER +
                    (colBase + t * 16 + r)] = (_Float16)(s * u);
            }
        }
    }
}

// ---------------- GEMM2: down projection, f16 WMMA ----------------
// grid: (8, MTILES), block: 128. Wave computes a 32x64 output tile.
__global__ __launch_bounds__(128)
void moe_gemm2(const _Float16* __restrict__ act, const _Float16* __restrict__ Wd,
               const int* __restrict__ offs, const int* __restrict__ counts,
               float* __restrict__ y) {
    const int rowBase = blockIdx.y * 32;
    if (rowBase >= offs[NE]) return;
    int e = 0;
    #pragma unroll
    for (int i = 1; i < NE; ++i) if (rowBase >= offs[i]) e = i;
    if (rowBase >= offs[e] + counts[e]) return;

    const int wave = threadIdx.x >> 5;
    const int lane = threadIdx.x & 31;
    const int half = lane >> 4;
    const int r    = lane & 15;
    const int colBase = blockIdx.x * 256 + wave * 64;

    const _Float16* aRow0 = act + (size_t)(rowBase + r) * INTER;
    const _Float16* aRow1 = act + (size_t)(rowBase + 16 + r) * INTER;

    const _Float16* b[4];
    #pragma unroll
    for (int t = 0; t < 4; ++t)
        b[t] = Wd + ((size_t)e * HID + (colBase + t * 16 + r)) * INTER + half * 8;

    v8f acc[2][4] = {};
    for (int kk = 0; kk < INTER; kk += 32) {
        v16h A0 = ld_frag(aRow0 + kk + half * 8, aRow0 + kk + 16 + half * 8);
        v16h A1 = ld_frag(aRow1 + kk + half * 8, aRow1 + kk + 16 + half * 8);
        #pragma unroll
        for (int t = 0; t < 4; ++t) {
            v16h B = ld_frag(b[t] + kk, b[t] + kk + 16);
            acc[0][t] = __builtin_amdgcn_wmma_f32_16x16x32_f16(false, A0, false, B,
                                                               (short)0, acc[0][t], false, false);
            acc[1][t] = __builtin_amdgcn_wmma_f32_16x16x32_f16(false, A1, false, B,
                                                               (short)0, acc[1][t], false, false);
        }
    }
    #pragma unroll
    for (int m = 0; m < 2; ++m)
        #pragma unroll
        for (int t = 0; t < 4; ++t)
            #pragma unroll
            for (int j = 0; j < 8; ++j)
                y[(size_t)(rowBase + m * 16 + j + half * 8) * HID +
                  (colBase + t * 16 + r)] = acc[m][t][j];
}

// ---------------- combine: out[n] = sum_k w[n,k] * y[pair_row[n,k]] ----------------
__global__ void k_combine(const float* __restrict__ y, const int* __restrict__ topk,
                          const float* __restrict__ topw, const int* __restrict__ pair_row,
                          float* __restrict__ out) {
    int i = blockIdx.x * blockDim.x + threadIdx.x;    // over N * H/4
    if (i >= N_TOK * (HID / 4)) return;
    int n  = i / (HID / 4);
    int hv = i % (HID / 4);
    float ax = 0.f, ay = 0.f, az = 0.f, aw = 0.f;
    #pragma unroll
    for (int k = 0; k < TOPK; ++k) {
        int e = topk[n * TOPK + k];
        float w = (e < 0 || e >= NE) ? 0.f : topw[n * TOPK + k];
        int row = pair_row[n * TOPK + k];
        float4 v = *(const float4*)(y + (size_t)row * HID + hv * 4);
        ax += w * v.x; ay += w * v.y; az += w * v.z; aw += w * v.w;
    }
    float4 o; o.x = ax; o.y = ay; o.z = az; o.w = aw;
    *(float4*)(out + (size_t)n * HID + hv * 4) = o;
}

// ---------------- workspace layout (bytes) ----------------
static constexpr size_t XH_OFF   = 0;                                   // N*H f16
static constexpr size_t XH_SZ    = (size_t)N_TOK * HID * 2;
static constexpr size_t WGU_OFF  = XH_OFF + XH_SZ;                      // E*2I*H f16
static constexpr size_t WGU_SZ   = (size_t)NE * 2 * INTER * HID * 2;
static constexpr size_t WD_OFF   = WGU_OFF + WGU_SZ;                    // E*H*I f16
static constexpr size_t WD_SZ    = (size_t)NE * HID * INTER * 2;
static constexpr size_t ACT_OFF  = WD_OFF + WD_SZ;                      // ROWCAP*I f16
static constexpr size_t ACT_SZ   = (size_t)ROWCAP * INTER * 2;
static constexpr size_t Y_OFF    = ACT_OFF + ACT_SZ;                    // ROWCAP*H f32
static constexpr size_t Y_SZ     = (size_t)ROWCAP * HID * 4;
static constexpr size_t CNT_OFF  = Y_OFF + Y_SZ;                        // ints, padded
static constexpr size_t CUR_OFF  = CNT_OFF + 256;
static constexpr size_t OFFS_OFF = CUR_OFF + 256;
static constexpr size_t SEG_OFF  = OFFS_OFF + 256;
static constexpr size_t PAIR_OFF = SEG_OFF + (size_t)ROWCAP * 4;

extern "C" void kernel_launch(void* const* d_in, const int* in_sizes, int n_in,
                              void* d_out, int out_size, void* d_ws, size_t ws_size,
                              hipStream_t stream) {
    const float* hs   = (const float*)d_in[0];
    const int*   topk = (const int*)  d_in[1];
    const float* topw = (const float*)d_in[2];
    const float* wgu  = (const float*)d_in[3];
    const float* wd   = (const float*)d_in[4];
    float* out = (float*)d_out;

    char* ws = (char*)d_ws;
    _Float16* Xh   = (_Float16*)(ws + XH_OFF);
    _Float16* Wgu  = (_Float16*)(ws + WGU_OFF);
    _Float16* Wd   = (_Float16*)(ws + WD_OFF);
    _Float16* act  = (_Float16*)(ws + ACT_OFF);
    float*    y    = (float*)   (ws + Y_OFF);
    int* counts    = (int*)(ws + CNT_OFF);
    int* cursors   = (int*)(ws + CUR_OFF);
    int* offs      = (int*)(ws + OFFS_OFF);
    int* seg       = (int*)(ws + SEG_OFF);
    int* pair_row  = (int*)(ws + PAIR_OFF);

    // 1) precision conversion fp32 -> f16 (streaming, one pass)
    {
        int n4;
        n4 = (N_TOK * HID) / 4;
        k_f32_to_f16<<<(n4 + 255) / 256, 256, 0, stream>>>(hs, Xh, n4);
        n4 = (NE * 2 * INTER * HID) / 4;
        k_f32_to_f16<<<(n4 + 255) / 256, 256, 0, stream>>>(wgu, Wgu, n4);
        n4 = (NE * HID * INTER) / 4;
        k_f32_to_f16<<<(n4 + 255) / 256, 256, 0, stream>>>(wd, Wd, n4);
    }

    // 2) routing
    k_zero_small<<<1, 64, 0, stream>>>(counts, cursors, offs);
    k_fill_seg<<<(ROWCAP + 255) / 256, 256, 0, stream>>>(seg);
    k_count<<<(N_TOK * TOPK + 255) / 256, 256, 0, stream>>>(topk, counts);
    k_scan<<<1, 1, 0, stream>>>(counts, offs);
    k_scatter<<<(N_TOK * TOPK + 255) / 256, 256, 0, stream>>>(topk, offs, cursors, seg, pair_row);

    // 3) gate_up GEMM + SiLU*up (routed rows only)
    moe_gemm1<<<dim3(3, MTILES), 128, 0, stream>>>(Xh, Wgu, seg, offs, counts, act);

    // 4) down GEMM
    moe_gemm2<<<dim3(8, MTILES), 128, 0, stream>>>(act, Wd, offs, counts, y);

    // 5) weighted top-k combine
    k_combine<<<(N_TOK * (HID / 4) + 255) / 256, 256, 0, stream>>>(y, topk, topw, pair_row, out);
}
